// ARLSTM_79027398246901
// MI455X (gfx1250) — matrix operand order repaired and checked by reading, not compiled, not run on hardware
//
#include <hip/hip_runtime.h>
#include <hip/hip_bf16.h>

// ---------------- problem constants ----------------
#define UNITS     1024
#define NF        128
#define BATCH     128
#define WARMUP_T  128
#define OUT_STEPS 48
#define KDIM      (NF + UNITS)      // 1152 : concatenated [x|h] reduction dim
#define KCH       (KDIM / 32)       // 36 K-chunks of 32 for the gate GEMM
#define KCH_D     (UNITS / 32)      // 32 K-chunks for the dense head
#define ZCOLS     (4 * UNITS)       // 4096 gate columns

typedef __bf16 bf16_t;
typedef __attribute__((ext_vector_type(16))) __bf16 v16bf;
typedef __attribute__((ext_vector_type(8)))  __bf16 v8bf;
typedef __attribute__((ext_vector_type(8)))  float  v8f;

// ---------------- small math helpers ----------------
__device__ __forceinline__ float fsigmoid(float x) {
  return 1.0f / (1.0f + __expf(-x));
}
__device__ __forceinline__ float ftanh(float x) {
  // overflow-safe tanh via exp of -2|x|
  float t = __expf(-2.0f * fabsf(x));
  float r = (1.0f - t) / (1.0f + t);
  return copysignf(r, x);
}

// Load a 16-bf16 A fragment for wmma_f32_16x16x32_bf16.
// 16-bit A layout (16x32): lane m (0-15)/m+16 holds row m;
// element e holds K = 16*hi + e (matches packed B).
__device__ __forceinline__ v16bf load_a16(const bf16_t* __restrict__ row, int hi) {
  v8bf lo = *(const v8bf*)(row + 16 * hi);
  v8bf hv = *(const v8bf*)(row + 16 * hi + 8);
  v16bf a;
#pragma unroll
  for (int i = 0; i < 8; ++i) { a[i] = lo[i]; a[8 + i] = hv[i]; }
  return a;
}

// Load a 16-bf16 B fragment from the pre-packed lane-major tile stream.
__device__ __forceinline__ v16bf load_b16(const bf16_t* __restrict__ p) {
  v8bf lo = *(const v8bf*)(p);
  v8bf hv = *(const v8bf*)(p + 8);
  v16bf b;
#pragma unroll
  for (int i = 0; i < 8; ++i) { b[i] = lo[i]; b[8 + i] = hv[i]; }
  return b;
}

// ---------------- one-time prep kernels ----------------
// Pack [Wx ; Wh] (1152 x 4096 fp32, row major) into bf16 WMMA-B tiles.
// B layout per 32x16 chunk (this code's contiguous k order, same as A): lane = hi*16 + n, element e -> (K = 16*hi + e, N = n).
// Packed index: (((nt*KCH + kc)*32 + lane)*16 + e)  -> each lane reads 32 contiguous bytes.
__global__ void ARLSTM_pack_wz(const float* __restrict__ Wx,
                               const float* __restrict__ Wh,
                               bf16_t* __restrict__ Wzp) {
  size_t tid = (size_t)blockIdx.x * blockDim.x + threadIdx.x;  // over 1152*4096
  int k = (int)(tid >> 12);
  int n = (int)(tid & 4095);
  float v = (k < NF) ? Wx[(size_t)k * ZCOLS + n]
                     : Wh[(size_t)(k - NF) * ZCOLS + n];
  int nt = n >> 4, nl = n & 15;
  int kc = k >> 5, kr = k & 31, hi = kr >> 4, e = kr & 15;
  size_t dst = (((size_t)(nt * KCH + kc) * 32) + hi * 16 + nl) * 16 + e;
  Wzp[dst] = (bf16_t)v;
}

// Pack Wd (1024 x 128) the same way.
__global__ void ARLSTM_pack_wd(const float* __restrict__ Wd,
                               bf16_t* __restrict__ Wdp) {
  size_t tid = (size_t)blockIdx.x * blockDim.x + threadIdx.x;  // over 1024*128
  int k = (int)(tid >> 7);
  int n = (int)(tid & 127);
  float v = Wd[(size_t)k * NF + n];
  int nt = n >> 4, nl = n & 15;
  int kc = k >> 5, kr = k & 31, hi = kr >> 4, e = kr & 15;
  size_t dst = (((size_t)(nt * KCH_D + kc) * 32) + hi * 16 + nl) * 16 + e;
  Wdp[dst] = (bf16_t)v;
}

// x[b][t][f] fp32 -> per-step bf16 tiles xbf[t][b][f]
__global__ void ARLSTM_convert_x(const float* __restrict__ x,
                                 bf16_t* __restrict__ xbf) {
  size_t tid = (size_t)blockIdx.x * blockDim.x + threadIdx.x;  // 128*128*128
  int b = (int)(tid >> 14);
  int t = (int)((tid >> 7) & 127);
  int f = (int)(tid & 127);
  xbf[(size_t)t * (BATCH * NF) + b * NF + f] = (bf16_t)x[tid];
}

__global__ void ARLSTM_init_state(bf16_t* __restrict__ h0,
                                  float* __restrict__ c) {
  size_t tid = (size_t)blockIdx.x * blockDim.x + threadIdx.x;  // 128*1024
  h0[tid] = (bf16_t)0.0f;
  c[tid] = 0.0f;
}

// ---------------- recurrent cell step ----------------
// Grid: (8 row-tiles, 64 unit-tiles). Block: 128 threads = 4 waves.
// Wave g computes the 16x16 z-tile of gate g for this (row, unit) tile via
// 36 x v_wmma_f32_16x16x32_bf16, then gates are fused through LDS.
__global__ __launch_bounds__(128)
void ARLSTM_gate_step(const bf16_t* __restrict__ Ax, int axStride,  // [128 x NF]
                      const bf16_t* __restrict__ Ah,                // [128 x UNITS]
                      const bf16_t* __restrict__ Wzp,
                      const float* __restrict__ bias,               // [4096]
                      float* __restrict__ c,                        // [128 x UNITS]
                      bf16_t* __restrict__ hOut) {                  // [128 x UNITS]
  const int wave = threadIdx.x >> 5;  // gate id 0..3 (i,f,g,o)
  const int lane = threadIdx.x & 31;
  const int hi   = lane >> 4;
  const int nl   = lane & 15;
  const int m0   = blockIdx.x * 16;   // batch-row tile
  const int ut   = blockIdx.y;        // unit tile 0..63
  const int n0   = ut * 16;
  const int ntg  = wave * 64 + ut;    // global z column-tile (gate offset folded in)

  const bf16_t* __restrict__ bline = Wzp + (size_t)(ntg * KCH) * 512 + lane * 16;
  const bf16_t* __restrict__ axRow = Ax + (size_t)(m0 + nl) * axStride;
  const bf16_t* __restrict__ ahRow = Ah + (size_t)(m0 + nl) * UNITS;

  v8f acc = {};
#pragma unroll 4
  for (int kc = 0; kc < KCH; ++kc) {
    const bf16_t* arow = (kc < 4) ? (axRow + kc * 32) : (ahRow + (kc - 4) * 32);
    v16bf a = load_a16(arow, hi);
    v16bf b = load_b16(bline);
    __builtin_prefetch(bline + 512, 0, 1);  // global_prefetch for next B chunk
    bline += 512;
    acc = __builtin_amdgcn_wmma_f32_16x16x32_bf16(
        false, a, false, b, (short)0, acc, false, false);
  }

  const float bsc = bias[wave * UNITS + n0 + nl];

  __shared__ float zlds[4][8][32];
#pragma unroll
  for (int v = 0; v < 8; ++v) zlds[wave][v][lane] = acc[v] + bsc;
  __syncthreads();

  // Fused gate math: wave w handles D-layout VGPR rows {2w, 2w+1}.
#pragma unroll
  for (int q = 0; q < 2; ++q) {
    const int v  = wave * 2 + q;
    const float zi = zlds[0][v][lane];
    const float zf = zlds[1][v][lane];
    const float zg = zlds[2][v][lane];
    const float zo = zlds[3][v][lane];
    const int M = m0 + v + 8 * hi;       // D layout: M = vgpr + 8*(lane>=16)
    const size_t idx = (size_t)M * UNITS + (n0 + nl);
    const float cv = c[idx];
    const float cn = fsigmoid(zf) * cv + fsigmoid(zi) * ftanh(zg);
    const float hn = fsigmoid(zo) * ftanh(cn);
    c[idx]    = cn;
    hOut[idx] = (bf16_t)hn;
  }
}

// ---------------- dense prediction head ----------------
// pred = h @ Wd + bd.  Grid: (8 row-tiles, 2). Block: 4 waves; wave w owns
// output column-tile nt = blockIdx.y*4 + w.
__global__ __launch_bounds__(128)
void ARLSTM_pred_step(const bf16_t* __restrict__ h,    // [128 x UNITS]
                      const bf16_t* __restrict__ Wdp,
                      const float* __restrict__ bd,    // [NF]
                      float* __restrict__ out,         // [B, OUT_STEPS, NF]
                      int step,
                      bf16_t* __restrict__ pOut) {     // [128 x NF] bf16 feedback
  const int wave = threadIdx.x >> 5;
  const int lane = threadIdx.x & 31;
  const int hi   = lane >> 4;
  const int nl   = lane & 15;
  const int m0   = blockIdx.x * 16;
  const int nt   = blockIdx.y * 4 + wave;  // 0..7

  const bf16_t* __restrict__ bline = Wdp + (size_t)(nt * KCH_D) * 512 + lane * 16;
  const bf16_t* __restrict__ hRow  = h + (size_t)(m0 + nl) * UNITS;

  v8f acc = {};
#pragma unroll 4
  for (int kc = 0; kc < KCH_D; ++kc) {
    v16bf a = load_a16(hRow + kc * 32, hi);
    v16bf b = load_b16(bline);
    bline += 512;
    acc = __builtin_amdgcn_wmma_f32_16x16x32_bf16(
        false, a, false, b, (short)0, acc, false, false);
  }

  const float bsc = bd[nt * 16 + nl];
#pragma unroll
  for (int v = 0; v < 8; ++v) {
    const int M = m0 + v + 8 * hi;
    const float val = acc[v] + bsc;
    out[(size_t)M * (OUT_STEPS * NF) + (size_t)step * NF + nt * 16 + nl] = val;
    pOut[(size_t)M * NF + nt * 16 + nl] = (bf16_t)val;
  }
}

// ---------------- host-side orchestration ----------------
extern "C" void kernel_launch(void* const* d_in, const int* in_sizes, int n_in,
                              void* d_out, int out_size, void* d_ws, size_t ws_size,
                              hipStream_t stream) {
  const float* x  = (const float*)d_in[0];   // [128,128,128]
  const float* Wx = (const float*)d_in[1];   // [128,4096]
  const float* Wh = (const float*)d_in[2];   // [1024,4096]
  const float* b  = (const float*)d_in[3];   // [4096]
  const float* Wd = (const float*)d_in[4];   // [1024,128]
  const float* bd = (const float*)d_in[5];   // [128]
  float* out = (float*)d_out;                // [128,48,128]

  // workspace carve-up (256B aligned), ~15.0 MB total
  char* ws = (char*)d_ws;
  size_t off = 0;
  auto alloc = [&](size_t bytes) {
    char* p = ws + off;
    off += (bytes + 255) & ~(size_t)255;
    return p;
  };
  bf16_t* Wzp = (bf16_t*)alloc(sizeof(bf16_t) * (size_t)KDIM * ZCOLS);       // 9.44 MB
  bf16_t* Wdp = (bf16_t*)alloc(sizeof(bf16_t) * (size_t)UNITS * NF);         // 256 KB
  bf16_t* xbf = (bf16_t*)alloc(sizeof(bf16_t) * (size_t)WARMUP_T * BATCH * NF); // 4 MB
  bf16_t* hA  = (bf16_t*)alloc(sizeof(bf16_t) * (size_t)BATCH * UNITS);      // 256 KB
  bf16_t* hB  = (bf16_t*)alloc(sizeof(bf16_t) * (size_t)BATCH * UNITS);      // 256 KB
  bf16_t* pA  = (bf16_t*)alloc(sizeof(bf16_t) * (size_t)BATCH * NF);         // 32 KB
  bf16_t* pB  = (bf16_t*)alloc(sizeof(bf16_t) * (size_t)BATCH * NF);         // 32 KB
  float*  c   = (float*) alloc(sizeof(float)  * (size_t)BATCH * UNITS);      // 512 KB
  (void)ws_size; (void)in_sizes; (void)n_in; (void)out_size;

  // one-time packing / init
  ARLSTM_pack_wz<<<(KDIM * ZCOLS) / 256, 256, 0, stream>>>(Wx, Wh, Wzp);
  ARLSTM_pack_wd<<<(UNITS * NF) / 256, 256, 0, stream>>>(Wd, Wdp);
  ARLSTM_convert_x<<<(BATCH * WARMUP_T * NF) / 256, 256, 0, stream>>>(x, xbf);
  ARLSTM_init_state<<<(BATCH * UNITS) / 256, 256, 0, stream>>>(hA, c);

  dim3 gateGrid(8, 64), blk(128);
  dim3 predGrid(8, 2);

  bf16_t* hc = hA; bf16_t* hn = hB;

  // warmup over the input sequence
  for (int t = 0; t < WARMUP_T; ++t) {
    ARLSTM_gate_step<<<gateGrid, blk, 0, stream>>>(
        xbf + (size_t)t * BATCH * NF, NF, hc, Wzp, b, c, hn);
    bf16_t* tmp = hc; hc = hn; hn = tmp;
  }

  // first prediction
  bf16_t* pc = pA; bf16_t* pn = pB;
  ARLSTM_pred_step<<<predGrid, blk, 0, stream>>>(hc, Wdp, bd, out, 0, pc);

  // autoregressive decode
  for (int s = 1; s < OUT_STEPS; ++s) {
    ARLSTM_gate_step<<<gateGrid, blk, 0, stream>>>(pc, NF, hc, Wzp, b, c, hn);
    bf16_t* tmp = hc; hc = hn; hn = tmp;
    ARLSTM_pred_step<<<predGrid, blk, 0, stream>>>(hc, Wdp, bd, out, s, pn);
    bf16_t* tmp2 = pc; pc = pn; pn = tmp2;
  }
}
